// STNRNN_4286377361418
// MI455X (gfx1250) — matrix operand order repaired and checked by reading, not compiled, not run on hardware
//
#include <hip/hip_runtime.h>
#include <hip/hip_bf16.h>

// STNRNN forward-Euler with K=1.0  =>  h_{t+1} = tanh(x_t*W_ih^T + b + h_t W_hh^T)
// B=32, T=8192, I=1, H=128. Serial over T; batch split across blocks.

typedef __attribute__((ext_vector_type(16))) __bf16        v16bf;
typedef __attribute__((ext_vector_type(8)))  float         v8f;
typedef __attribute__((ext_vector_type(4)))  float         f32x4;
typedef __attribute__((ext_vector_type(4)))  unsigned int  u32x4;

#define HDIM       128
#define T_LEN      8192
#define B_TOT      32
#define ROWS       16          // batch rows per block
#define ROW_STRIDE 272         // bytes per LDS h row: 128*2 + 16 pad (bank skew)
#define XCHUNK     32          // timesteps of x staged per refill

__device__ __forceinline__ float tanh_fast(float x) {
#if __has_builtin(__builtin_amdgcn_tanhf)
  return __builtin_amdgcn_tanhf(x);
#elif __has_builtin(__builtin_amdgcn_tanh_f32)
  return __builtin_amdgcn_tanh_f32(x);
#else
  float y;
  asm volatile("v_tanh_f32 %0, %1" : "=v"(y) : "v"(x));
  return y;
#endif
}

__global__ __launch_bounds__(256, 1)
void stnrnn_kernel(const float* __restrict__ x,      // [B, T, 1]
                   const float* __restrict__ W_ih,   // [H, 1]
                   const float* __restrict__ W_hh,   // [H, H]
                   const float* __restrict__ b_ih,   // [H]
                   const float* __restrict__ b_hh,   // [H]
                   float* __restrict__ out)          // [B,T,H] ++ [1,B,H]
{
  __shared__ __align__(16) unsigned char hbuf[2][ROWS * ROW_STRIDE]; // bf16 h, double-buffered
  __shared__ __align__(16) float xstage[XCHUNK][ROWS];               // x staged, transposed

  const int tid   = threadIdx.x;
  const int lane  = tid & 31;
  const int wid   = tid >> 5;         // 0..7  -> N-tile owned by this wave
  const int half  = lane >> 4;        // 0/1
  const int l16   = lane & 15;
  const int b0    = blockIdx.x * ROWS;

  const int nglob = wid * 16 + l16;   // hidden column this lane owns (C/D layout)
  const int mbase = half * 8;         // first of the 8 batch rows this lane owns (C/D)

  // ---- B operand: B[k][n] = W_hh[n][k] (i.e. h @ W_hh^T), bf16, 4 K-chunks of 32 ----
  // 16-bit B layout (mirror of A): lane = half*16 + n,
  // element e holds k_local = half*8 + e (e<8), 16 + half*8 + (e-8) (e>=8)
  v16bf Bm[4];
  {
    const int klo = half * 8;
    #pragma unroll
    for (int c = 0; c < 4; ++c) {
      const float* wr = W_hh + (size_t)nglob * HDIM + c * 32 + klo;
      #pragma unroll
      for (int e = 0; e < 16; ++e) Bm[c][e] = (__bf16)wr[e < 8 ? e : e + 8];
    }
  }
  const float wihn  = W_ih[nglob];                 // I == 1
  const float biasn = b_ih[nglob] + b_hh[nglob];

  // ---- h0 = 0: zero LDS buffer 0 ----
  for (int i = tid; i < (ROWS * ROW_STRIDE) / 4; i += (int)blockDim.x)
    ((unsigned int*)hbuf[0])[i] = 0u;
  __syncthreads();

  const size_t TH       = (size_t)T_LEN * HDIM;
  const size_t rowbase0 = (size_t)(b0 + mbase) * TH + nglob;
  const int    aoff     = half * 16;               // byte offset of this half's K range

  float hc[8];
  #pragma unroll 1
  for (int t = 0; t < T_LEN; ++t) {
    const int tt = t & (XCHUNK - 1);
    if (tt == 0) {
      // stage x[b0..b0+15][t..t+31] into LDS, transposed to xstage[tt][m]
      if (tid < 128) {
        const int m = tid >> 3, s = tid & 7;
        f32x4 xv = *(const f32x4*)(x + (size_t)(b0 + m) * T_LEN + t + s * 4);
        #pragma unroll
        for (int j = 0; j < 4; ++j) xstage[s * 4 + j][m] = xv[j];
      }
      __syncthreads();
    }

    const int p = t & 1;

    // ---- A operand: h_t from LDS, documented 16-bit 16x32 A layout per chunk:
    // lane(<16) row=l16 reads K e=0..7 at +0 and e=16..23 at +32; lane(>=16) +16 shift.
    v16bf Am[4];
    #pragma unroll
    for (int c = 0; c < 4; ++c) {
      const unsigned char* rb = hbuf[p] + (size_t)l16 * ROW_STRIDE + c * 64 + aoff;
      union { v16bf v; u32x4 q[2]; } u;
      u.q[0] = *(const u32x4*)(rb);
      u.q[1] = *(const u32x4*)(rb + 32);
      Am[c] = u.v;
    }

    // ---- seed accumulator with xp = x[m]*W_ih[n] + (b_ih+b_hh)[n] ----
    const f32x4* xr = (const f32x4*)&xstage[tt][mbase];
    f32x4 x0 = xr[0], x1 = xr[1];
    v8f acc;
    #pragma unroll
    for (int r = 0; r < 4; ++r) acc[r]     = fmaf(x0[r], wihn, biasn);
    #pragma unroll
    for (int r = 0; r < 4; ++r) acc[4 + r] = fmaf(x1[r], wihn, biasn);

    // ---- h_t @ W_hh^T : 4 chained bf16 WMMAs over K=128 ----
    #pragma unroll
    for (int c = 0; c < 4; ++c)
      acc = __builtin_amdgcn_wmma_f32_16x16x32_bf16(false, Am[c], false, Bm[c],
                                                    (short)0, acc, false, false);

    // ---- K == 1.0 => h_{t+1} = tanh(pre) ----
    #pragma unroll
    for (int r = 0; r < 8; ++r) hc[r] = tanh_fast(acc[r]);

    // ---- states[b, t, h] ----
    float* op = out + rowbase0 + (size_t)t * HDIM;
    #pragma unroll
    for (int r = 0; r < 8; ++r) op[(size_t)r * TH] = hc[r];

    // ---- h_{t+1} -> LDS (bf16, row-major) for next step's A loads ----
    unsigned char* wb = hbuf[p ^ 1];
    #pragma unroll
    for (int r = 0; r < 8; ++r)
      *(__bf16*)(wb + (size_t)(mbase + r) * ROW_STRIDE + (size_t)nglob * 2) = (__bf16)hc[r];

    __syncthreads();
  }

  // ---- h_last = states[:, T-1, :] (still live in hc) ----
  float* hl = out + (size_t)B_TOT * TH + (size_t)(b0 + mbase) * HDIM + nglob;
  #pragma unroll
  for (int r = 0; r < 8; ++r) hl[(size_t)r * HDIM] = hc[r];
}

extern "C" void kernel_launch(void* const* d_in, const int* in_sizes, int n_in,
                              void* d_out, int out_size, void* d_ws, size_t ws_size,
                              hipStream_t stream) {
  (void)in_sizes; (void)n_in; (void)out_size; (void)d_ws; (void)ws_size;
  const float* x    = (const float*)d_in[0];
  const float* W_ih = (const float*)d_in[1];
  const float* W_hh = (const float*)d_in[2];
  const float* b_ih = (const float*)d_in[3];
  const float* b_hh = (const float*)d_in[4];
  float* out = (float*)d_out;

  dim3 grid(B_TOT / ROWS);   // 2 blocks of 16 batch rows
  dim3 block(256);           // 8 wave32s, one 16-col N-tile each
  stnrnn_kernel<<<grid, block, 0, stream>>>(x, W_ih, W_hh, b_ih, b_hh, out);
}
